// DeepLSTMDecoderLayer_5712306504123
// MI455X (gfx1250) — matrix operand order repaired and checked by reading, not compiled, not run on hardware
//
#include <hip/hip_runtime.h>
#include <hip/hip_bf16.h>

typedef __attribute__((ext_vector_type(16))) __bf16 v16bf;
typedef __attribute__((ext_vector_type(8)))  __bf16 v8bf;
typedef __attribute__((ext_vector_type(8)))  float  v8f;

#define T_LEN   128
#define B_LEN   64
#define H_LEN   1024
#define S_LEN   1024
#define HEADS   16
#define HD      64
#define FILTER  4096

// build a 16-wide bf16 fragment from two contiguous 16-byte chunks
__device__ __forceinline__ v16bf ld_frag(const __bf16* p0, const __bf16* p1) {
  v8bf lo = *(const v8bf*)p0;
  v8bf hi = *(const v8bf*)p1;
  return __builtin_shufflevector(lo, hi, 0, 1, 2, 3, 4, 5, 6, 7,
                                 8, 9, 10, 11, 12, 13, 14, 15);
}

// ---------------------------------------------------------------------------
// small elementwise kernels
// ---------------------------------------------------------------------------
__global__ void cast_f32_to_bf16(const float* __restrict__ s, __bf16* __restrict__ d, size_t n) {
  size_t i = (size_t)blockIdx.x * 256 + threadIdx.x;
  if (i < n) d[i] = (__bf16)s[i];
}

// pack x (T*B, H) fp32 into left half of xc (T*B, 2H) bf16
__global__ void cast_x_to_xc(const float* __restrict__ x, __bf16* __restrict__ xc, size_t n) {
  size_t i = (size_t)blockIdx.x * 256 + threadIdx.x;
  if (i < n) {
    size_t r = i >> 10, c = i & 1023;
    xc[r * 2048 + c] = (__bf16)x[i];
  }
}

__global__ void init_state(float* c, float* h, __bf16* hbf, int n) {
  int i = blockIdx.x * 256 + threadIdx.x;
  if (i < n) { c[i] = 0.f; h[i] = 0.f; hbf[i] = (__bf16)0.f; }
}

__global__ void copy_out_state(const float* c, const float* h, float* out, int n) {
  int i = blockIdx.x * 256 + threadIdx.x;
  if (i < n) { out[i] = c[i]; out[n + i] = h[i]; }
}

// ---------------------------------------------------------------------------
// bf16 WMMA GEMM:  C[M,N] = act( post_scale*(A@B + bias) + add32 )
// A row-major [M,K] bf16 (lda), B row-major [K,N] bf16 (ldb)
// block tile 128x128, 8 waves (4x2), wave tile 32x64 (2x4 of 16x16)
// A staged row-major (K contiguous); B staged TRANSPOSED (K contiguous)
// ---------------------------------------------------------------------------
#define BSTR 40   // padded K-stride of transposed B tile (multiple of 8 for b128)

__global__ __launch_bounds__(256) void gemm_bf16(
    const __bf16* __restrict__ A, int lda,
    const __bf16* __restrict__ Bm, int ldb,
    const float* __restrict__ bias,
    const float* __restrict__ add32,   // optional fp32 addend [M,ldc]
    float* __restrict__ C32, __bf16* __restrict__ C16, int ldc,
    int M, int N, int K, float post_scale, int relu)
{
  __shared__ __align__(64) __bf16 As[128 * 32];
  __shared__ __align__(64) __bf16 Bs[128 * BSTR];

  const int tid = threadIdx.x;
  const int lane = tid & 31;
  const int wid = tid >> 5;
  const int wm = wid >> 1, wn = wid & 1;
  const int g = lane >> 4, lc = lane & 15;
  const int m0 = blockIdx.y * 128, n0 = blockIdx.x * 128;

  v8f acc[2][4] = {};

  const int arow = tid >> 1;          // 0..127
  const int acol = (tid & 1) * 16;    // 0 / 16
  const int brow = tid >> 3;          // 0..31  (k index)
  const int bcol = (tid & 7) * 16;    // 0..112 (n index)

  // branch-free staging: clamp OOB rows (epilogue guards stores)
  const int gr = (m0 + arow < M) ? (m0 + arow) : (M - 1);
  const __bf16* apb = A + (size_t)gr * lda + acol;
  const __bf16* bpb = Bm + (size_t)brow * ldb + n0 + bcol;

  for (int k0 = 0; k0 < K; k0 += 32) {
    // ---- stage A (row-major, 16B vector ops) ----
    {
      const __bf16* ap = apb + k0;
      *(v8bf*)&As[arow * 32 + acol]     = *(const v8bf*)ap;
      *(v8bf*)&As[arow * 32 + acol + 8] = *(const v8bf*)(ap + 8);
      if (k0 + 32 < K) __builtin_prefetch(ap + 32);
    }
    // ---- stage B transposed: Bs[n*BSTR + k] ----
    {
      const __bf16* bp = bpb + (size_t)k0 * ldb;
      v8bf b0 = *(const v8bf*)bp;
      v8bf b1 = *(const v8bf*)(bp + 8);
#pragma unroll
      for (int i = 0; i < 8; ++i) {
        Bs[(bcol + i) * BSTR + brow]     = b0[i];
        Bs[(bcol + 8 + i) * BSTR + brow] = b1[i];
      }
      if (k0 + 32 < K) __builtin_prefetch(bp + (size_t)32 * ldb);
    }
    __syncthreads();

    v16bf af[2], bfr[4];
#pragma unroll
    for (int mt = 0; mt < 2; ++mt) {
      const __bf16* p = &As[(wm * 32 + mt * 16 + lc) * 32 + g * 8];
      af[mt] = ld_frag(p, p + 16);           // k = g*8+e | 16+g*8+e
    }
#pragma unroll
    for (int nt = 0; nt < 4; ++nt) {
      const __bf16* p = &Bs[(wn * 64 + nt * 16 + lc) * BSTR + g * 8];
      bfr[nt] = ld_frag(p, p + 16);          // k = g*8+e | 16+g*8+e
    }
#pragma unroll
    for (int mt = 0; mt < 2; ++mt)
#pragma unroll
      for (int nt = 0; nt < 4; ++nt)
        acc[mt][nt] = __builtin_amdgcn_wmma_f32_16x16x32_bf16(
            false, af[mt], false, bfr[nt], (short)0, acc[mt][nt], false, false);
    __syncthreads();
  }

#pragma unroll
  for (int mt = 0; mt < 2; ++mt)
#pragma unroll
    for (int nt = 0; nt < 4; ++nt)
#pragma unroll
      for (int i = 0; i < 8; ++i) {
        int r  = m0 + wm * 32 + mt * 16 + g * 8 + i;
        int cN = n0 + wn * 64 + nt * 16 + lc;
        if (r < M) {
          float v = acc[mt][nt][i];
          if (bias) v += bias[cN];
          v *= post_scale;
          if (add32) v += add32[(size_t)r * ldc + cN];
          if (relu)  v = fmaxf(v, 0.f);
          if (C32) C32[(size_t)r * ldc + cN] = v;
          if (C16) C16[(size_t)r * ldc + cN] = (__bf16)v;
        }
      }
}

// ---------------------------------------------------------------------------
// flash-style attention, one block per (b, head), 8 waves x 16 query rows
// Q/K/V token-major: [(t_or_s)*B + b][H], head slice h*HD..+64
// K chunk staged row-major (d contiguous), V chunk staged transposed
// bias_mode 0: bias[b*Slen + s] | 1: bias[t*Slen + s]
// ---------------------------------------------------------------------------
#define KSTR 72   // Ks row stride (elems)
#define VSTR 40   // Vs row stride (elems)

__global__ __launch_bounds__(256) void attn_bf16(
    const __bf16* __restrict__ Q, const __bf16* __restrict__ Kb,
    const __bf16* __restrict__ Vb, const float* __restrict__ bias,
    int bias_mode, float* __restrict__ ctx, int accumulate, int Slen)
{
  const int bh = blockIdx.x;
  const int b = bh / HEADS, h = bh % HEADS;
  const int tid = threadIdx.x;
  const int wid = tid >> 5, lane = tid & 31;
  const int g = lane >> 4, lc = lane & 15;
  const int t0 = wid * 16;

  __shared__ __align__(64) __bf16 Ks[32 * KSTR];
  __shared__ __align__(64) __bf16 Vs[64 * VSTR];
  __shared__ __align__(64) __bf16 Pbuf[8][16 * 32];

  // staging assignment: 8 elems per thread per tile
  const int srow = tid >> 3;          // 0..31 (key within chunk)
  const int scol = (tid & 7) * 8;     // 0..56 (head-dim offset)

  // Q A-fragments (K depth 64 -> 2 frags of 32), vector loads
  const __bf16* qp = Q + ((size_t)(t0 + lc) * B_LEN + b) * H_LEN + h * HD;
  v16bf aq[2];
#pragma unroll
  for (int f = 0; f < 2; ++f)
    aq[f] = ld_frag(qp + 32 * f + g * 8, qp + 32 * f + 16 + g * 8);

  float m[8], l[8];
  v8f oacc[4] = {};
#pragma unroll
  for (int i = 0; i < 8; ++i) { m[i] = -1e30f; l[i] = 0.f; }

  for (int s0 = 0; s0 < Slen; s0 += 32) {
    // ---- stage K row-major, V transposed ----
    {
      const __bf16* kp = Kb + ((size_t)(s0 + srow) * B_LEN + b) * H_LEN + h * HD + scol;
      *(v8bf*)&Ks[srow * KSTR + scol] = *(const v8bf*)kp;
      const __bf16* vp = Vb + ((size_t)(s0 + srow) * B_LEN + b) * H_LEN + h * HD + scol;
      v8bf vv = *(const v8bf*)vp;
#pragma unroll
      for (int i = 0; i < 8; ++i) Vs[(scol + i) * VSTR + srow] = vv[i];
    }
    __syncthreads();

    // ---- logits: two 16x16 subtiles over 32 keys ----
    v8f st[2];
#pragma unroll
    for (int j = 0; j < 2; ++j) {
      v8f c = {};
#pragma unroll
      for (int f = 0; f < 2; ++f) {
        const __bf16* p = &Ks[(j * 16 + lc) * KSTR + 32 * f + g * 8];
        v16bf bk = ld_frag(p, p + 16);
        c = __builtin_amdgcn_wmma_f32_16x16x32_bf16(false, aq[f], false, bk,
                                                    (short)0, c, false, false);
      }
#pragma unroll
      for (int i = 0; i < 8; ++i) {
        int t = t0 + g * 8 + i;
        int s = s0 + j * 16 + lc;
        c[i] += (bias_mode == 0) ? bias[(size_t)b * Slen + s]
                                 : bias[(size_t)t * Slen + s];
      }
      st[j] = c;
    }
    // ---- online softmax (row reductions within 16-lane half) ----
    float nm[8];
#pragma unroll
    for (int i = 0; i < 8; ++i) {
      float mx = fmaxf(st[0][i], st[1][i]);
      for (int off = 1; off < 16; off <<= 1)
        mx = fmaxf(mx, __shfl_xor(mx, off));
      nm[i] = fmaxf(m[i], mx);
    }
#pragma unroll
    for (int j = 0; j < 2; ++j)
#pragma unroll
      for (int i = 0; i < 8; ++i)
        st[j][i] = __expf(st[j][i] - nm[i]);
#pragma unroll
    for (int i = 0; i < 8; ++i) {
      float sm = st[0][i] + st[1][i];
      for (int off = 1; off < 16; off <<= 1)
        sm += __shfl_xor(sm, off);
      float alpha = __expf(m[i] - nm[i]);
      l[i] = l[i] * alpha + sm;
      m[i] = nm[i];
#pragma unroll
      for (int nt = 0; nt < 4; ++nt) oacc[nt][i] *= alpha;
    }
    // ---- P: C layout -> row-major LDS -> A fragment ----
#pragma unroll
    for (int j = 0; j < 2; ++j)
#pragma unroll
      for (int i = 0; i < 8; ++i)
        Pbuf[wid][(g * 8 + i) * 32 + j * 16 + lc] = (__bf16)st[j][i];
    __syncthreads();
    const __bf16* pp = &Pbuf[wid][lc * 32 + g * 8];
    v16bf ap = ld_frag(pp, pp + 16);
    // ---- P @ V ----
#pragma unroll
    for (int nt = 0; nt < 4; ++nt) {
      const __bf16* p = &Vs[(nt * 16 + lc) * VSTR + g * 8];
      v16bf bv = ld_frag(p, p + 16);
      oacc[nt] = __builtin_amdgcn_wmma_f32_16x16x32_bf16(false, ap, false, bv,
                                                         (short)0, oacc[nt], false, false);
    }
    __syncthreads();
  }

#pragma unroll
  for (int nt = 0; nt < 4; ++nt)
#pragma unroll
    for (int i = 0; i < 8; ++i) {
      int t = t0 + g * 8 + i;
      int d = h * HD + nt * 16 + lc;
      float v = oacc[nt][i] / l[i];
      size_t idx = ((size_t)t * B_LEN + b) * H_LEN + d;
      if (accumulate) ctx[idx] += v; else ctx[idx] = v;
    }
}

// ---------------------------------------------------------------------------
// fused per-gate LayerNorm + LSTM cell update; one block per batch row
// ---------------------------------------------------------------------------
__device__ __forceinline__ float sigf(float x) { return 1.f / (1.f + __expf(-x)); }

__global__ __launch_bounds__(1024) void lstm_step(
    const float* __restrict__ gates,        // [B, 4H]
    const float* __restrict__ ln_scale, const float* __restrict__ ln_bias,
    float* __restrict__ cbuf, float* __restrict__ hbuf,
    __bf16* __restrict__ h_bf, __bf16* __restrict__ hid_t)
{
  const int b = blockIdx.x, e = threadIdx.x;
  const int lane = e & 31, wv = e >> 5;
  const float* gr = gates + (size_t)b * (4 * H_LEN);

  float v[4];
#pragma unroll
  for (int gt = 0; gt < 4; ++gt) v[gt] = gr[gt * H_LEN + e];

  __shared__ float sred[4][2][32];
#pragma unroll
  for (int gt = 0; gt < 4; ++gt) {
    float s = v[gt], s2 = v[gt] * v[gt];
    for (int off = 1; off < 32; off <<= 1) {
      s  += __shfl_xor(s,  off);
      s2 += __shfl_xor(s2, off);
    }
    if (lane == 0) { sred[gt][0][wv] = s; sred[gt][1][wv] = s2; }
  }
  __syncthreads();
  if (wv == 0) {
#pragma unroll
    for (int gt = 0; gt < 4; ++gt) {
      float s = sred[gt][0][lane], s2 = sred[gt][1][lane];
      for (int off = 1; off < 32; off <<= 1) {
        s  += __shfl_xor(s,  off);
        s2 += __shfl_xor(s2, off);
      }
      if (lane == 0) { sred[gt][0][0] = s; sred[gt][1][0] = s2; }
    }
  }
  __syncthreads();

  float gn[4];
#pragma unroll
  for (int gt = 0; gt < 4; ++gt) {
    float mu  = sred[gt][0][0] * (1.f / H_LEN);
    float var = sred[gt][1][0] * (1.f / H_LEN) - mu * mu;
    gn[gt] = (v[gt] - mu) * rsqrtf(var + 1e-6f) * ln_scale[gt * H_LEN + e]
             + ln_bias[gt * H_LEN + e];
  }
  float c_old = cbuf[b * H_LEN + e];
  float nc = sigf(gn[2]) * c_old + sigf(gn[0]) * tanhf(gn[1]);
  float nh = sigf(gn[3]) * tanhf(nc);
  cbuf[b * H_LEN + e]  = nc;
  hbuf[b * H_LEN + e]  = nh;
  h_bf[b * H_LEN + e]  = (__bf16)nh;
  hid_t[b * H_LEN + e] = (__bf16)nh;
}

// ---------------------------------------------------------------------------
// host orchestration
// ---------------------------------------------------------------------------
extern "C" void kernel_launch(void* const* d_in, const int* in_sizes, int n_in,
                              void* d_out, int out_size, void* d_ws, size_t ws_size,
                              hipStream_t stream) {
  (void)in_sizes; (void)n_in; (void)out_size; (void)ws_size;
  const float* x        = (const float*)d_in[0];
  const float* memory   = (const float*)d_in[1];
  const float* src_bias = (const float*)d_in[2];
  const float* tgt_bias = (const float*)d_in[3];
  const float* wq = (const float*)d_in[4];  const float* bq = (const float*)d_in[5];
  const float* wk = (const float*)d_in[6];  const float* bk = (const float*)d_in[7];
  const float* wv = (const float*)d_in[8];  const float* bv = (const float*)d_in[9];
  const float* wo = (const float*)d_in[10]; const float* bo = (const float*)d_in[11];
  const float* w_gates = (const float*)d_in[12];
  const float* b_gates = (const float*)d_in[13];
  const float* ln_scale = (const float*)d_in[14];
  const float* ln_bias  = (const float*)d_in[15];
  const float* w1 = (const float*)d_in[16]; const float* b1 = (const float*)d_in[17];
  const float* w2 = (const float*)d_in[18]; const float* b2 = (const float*)d_in[19];

  const size_t TB = (size_t)T_LEN * B_LEN;            // 8192 tokens
  const size_t SB = (size_t)S_LEN * B_LEN;            // 65536 tokens

  size_t off = 0;
  auto alloc = [&](size_t bytes) {
    size_t p = off; off += (bytes + 255) & ~(size_t)255; return (char*)d_ws + p;
  };
  __bf16* mem_bf = (__bf16*)alloc(SB * H_LEN * 2);
  __bf16* wq_bf  = (__bf16*)alloc((size_t)H_LEN * H_LEN * 2);
  __bf16* wk_bf  = (__bf16*)alloc((size_t)H_LEN * H_LEN * 2);
  __bf16* wv_bf  = (__bf16*)alloc((size_t)H_LEN * H_LEN * 2);
  __bf16* wo_bf  = (__bf16*)alloc((size_t)H_LEN * H_LEN * 2);
  __bf16* wgx_bf = (__bf16*)alloc((size_t)2 * H_LEN * 4 * H_LEN * 2);
  __bf16* wgh_bf = (__bf16*)alloc((size_t)H_LEN * 4 * H_LEN * 2);
  __bf16* w1_bf  = (__bf16*)alloc((size_t)H_LEN * FILTER * 2);
  __bf16* w2_bf  = (__bf16*)alloc((size_t)FILTER * H_LEN * 2);
  __bf16* q_bf   = (__bf16*)alloc(TB * H_LEN * 2);
  __bf16* kx_bf  = (__bf16*)alloc(TB * H_LEN * 2);
  __bf16* vx_bf  = (__bf16*)alloc(TB * H_LEN * 2);
  __bf16* km_bf  = (__bf16*)alloc(SB * H_LEN * 2);
  __bf16* vm_bf  = (__bf16*)alloc(SB * H_LEN * 2);
  float*  ctx    = (float*) alloc(TB * H_LEN * 4);
  __bf16* ctx_bf = (__bf16*)alloc(TB * H_LEN * 2);
  __bf16* xc_bf  = (__bf16*)alloc(TB * 2 * H_LEN * 2);
  float*  pre    = (float*) alloc(TB * 4 * H_LEN * 4);
  float*  gates  = (float*) alloc((size_t)B_LEN * 4 * H_LEN * 4);
  float*  cbuf   = (float*) alloc((size_t)B_LEN * H_LEN * 4);
  float*  hbuf   = (float*) alloc((size_t)B_LEN * H_LEN * 4);
  __bf16* h_bf   = (__bf16*)alloc((size_t)B_LEN * H_LEN * 2);
  __bf16* hid_bf = (__bf16*)alloc(TB * H_LEN * 2);
  __bf16* act_bf = (__bf16*)alloc(TB * FILTER * 2);

  auto cast = [&](const float* s, __bf16* d, size_t n) {
    cast_f32_to_bf16<<<dim3((unsigned)((n + 255) / 256)), 256, 0, stream>>>(s, d, n);
  };
  auto gemm = [&](const __bf16* A, int lda, const __bf16* Bm, int ldb,
                  const float* bias, const float* add32,
                  float* C32, __bf16* C16, int ldc,
                  int M, int N, int K, float scale, int relu) {
    dim3 grid(N / 128, (M + 127) / 128);
    gemm_bf16<<<grid, 256, 0, stream>>>(A, lda, Bm, ldb, bias, add32,
                                        C32, C16, ldc, M, N, K, scale, relu);
  };

  // ---- casts ----
  cast_x_to_xc<<<dim3((unsigned)((TB * H_LEN + 255) / 256)), 256, 0, stream>>>(x, xc_bf, TB * H_LEN);
  cast(memory, mem_bf, SB * H_LEN);
  cast(wq, wq_bf, (size_t)H_LEN * H_LEN);
  cast(wk, wk_bf, (size_t)H_LEN * H_LEN);
  cast(wv, wv_bf, (size_t)H_LEN * H_LEN);
  cast(wo, wo_bf, (size_t)H_LEN * H_LEN);
  cast(w_gates, wgx_bf, (size_t)2 * H_LEN * 4 * H_LEN);
  cast(w_gates + (size_t)2 * H_LEN * 4 * H_LEN, wgh_bf, (size_t)H_LEN * 4 * H_LEN);
  cast(w1, w1_bf, (size_t)H_LEN * FILTER);
  cast(w2, w2_bf, (size_t)FILTER * H_LEN);

  // ---- projections (q scaled by HD^-0.5) ----
  gemm(xc_bf, 2 * H_LEN, wq_bf, H_LEN, bq, nullptr, nullptr, q_bf,  H_LEN, (int)TB, H_LEN, H_LEN, 0.125f, 0);
  gemm(xc_bf, 2 * H_LEN, wk_bf, H_LEN, bk, nullptr, nullptr, kx_bf, H_LEN, (int)TB, H_LEN, H_LEN, 1.f, 0);
  gemm(xc_bf, 2 * H_LEN, wv_bf, H_LEN, bv, nullptr, nullptr, vx_bf, H_LEN, (int)TB, H_LEN, H_LEN, 1.f, 0);
  gemm(mem_bf, H_LEN, wk_bf, H_LEN, bk, nullptr, nullptr, km_bf, H_LEN, (int)SB, H_LEN, H_LEN, 1.f, 0);
  gemm(mem_bf, H_LEN, wv_bf, H_LEN, bv, nullptr, nullptr, vm_bf, H_LEN, (int)SB, H_LEN, H_LEN, 1.f, 0);

  // ---- attention: cross writes ctx, self accumulates ----
  attn_bf16<<<B_LEN * HEADS, 256, 0, stream>>>(q_bf, km_bf, vm_bf, src_bias, 0, ctx, 0, S_LEN);
  attn_bf16<<<B_LEN * HEADS, 256, 0, stream>>>(q_bf, kx_bf, vx_bf, tgt_bias, 1, ctx, 1, T_LEN);

  // ---- output projection -> right half of xc ----
  cast(ctx, ctx_bf, TB * H_LEN);
  gemm(ctx_bf, H_LEN, wo_bf, H_LEN, bo, nullptr, nullptr, xc_bf + H_LEN, 2 * H_LEN,
       (int)TB, H_LEN, H_LEN, 1.f, 0);

  // ---- hoisted LSTM input GEMM: pre = xc @ Wgx + b_gates ----
  gemm(xc_bf, 2 * H_LEN, wgx_bf, 4 * H_LEN, b_gates, nullptr, pre, nullptr, 4 * H_LEN,
       (int)TB, 4 * H_LEN, 2 * H_LEN, 1.f, 0);

  // ---- recurrent scan ----
  init_state<<<(B_LEN * H_LEN + 255) / 256, 256, 0, stream>>>(cbuf, hbuf, h_bf, B_LEN * H_LEN);
  for (int t = 0; t < T_LEN; ++t) {
    gemm(h_bf, H_LEN, wgh_bf, 4 * H_LEN, nullptr,
         pre + (size_t)t * B_LEN * 4 * H_LEN,
         gates, nullptr, 4 * H_LEN, B_LEN, 4 * H_LEN, H_LEN, 1.f, 0);
    lstm_step<<<B_LEN, 1024, 0, stream>>>(gates, ln_scale, ln_bias, cbuf, hbuf,
                                          h_bf, hid_bf + (size_t)t * B_LEN * H_LEN);
  }

  // ---- FFN ----
  gemm(hid_bf, H_LEN, w1_bf, FILTER, b1, nullptr, nullptr, act_bf, FILTER,
       (int)TB, FILTER, H_LEN, 1.f, 1);
  gemm(act_bf, FILTER, w2_bf, H_LEN, b2, nullptr, (float*)d_out, nullptr, H_LEN,
       (int)TB, H_LEN, FILTER, 1.f, 0);

  // ---- cf, hf ----
  copy_out_state<<<(B_LEN * H_LEN + 255) / 256, 256, 0, stream>>>(
      cbuf, hbuf, (float*)d_out + TB * H_LEN, B_LEN * H_LEN);
}
